// BahdanauAttention_61074434949251
// MI455X (gfx1250) — compile-verified, matrix operand order checked by reading, not run on hardware
//
#include <hip/hip_runtime.h>
#include <hip/hip_bf16.h>

#define B_ 32
#define S_ 2048
#define H_ 1024
#define U_ 1024

typedef __attribute__((ext_vector_type(16))) __bf16 v16bf;
typedef __attribute__((ext_vector_type(8)))  float  v8f;

__device__ __forceinline__ unsigned short f32_to_bf16_rne(float f) {
  unsigned int u = __float_as_uint(f);
  unsigned int r = u + 0x7FFFu + ((u >> 16) & 1u);
  return (unsigned short)(r >> 16);
}

// ---------------- Kernel 1: W1 [H,U] f32 -> W1T [U,H] bf16 ----------------
__global__ void w1_transpose_bf16(const float* __restrict__ W1,
                                  unsigned short* __restrict__ W1T) {
  int idx = blockIdx.x * blockDim.x + threadIdx.x;   // 0 .. H*U-1
  int h = idx & (H_ - 1);
  int u = idx >> 10;
  W1T[u * H_ + h] = f32_to_bf16_rne(W1[h * U_ + u]);
}

// ---- Kernel 2: hb[b,u] = last_h[b,:] @ W2[:,u] + W2_b[u] + W1_b[u] -------
__global__ void combined_bias_kernel(const float* __restrict__ hN,
                                     const float* __restrict__ W2,
                                     const float* __restrict__ W2b,
                                     const float* __restrict__ W1b,
                                     float* __restrict__ hb) {
  int idx = blockIdx.x * blockDim.x + threadIdx.x;   // 0 .. B*U-1
  int u = idx & (U_ - 1);
  int b = idx >> 10;
  const float* hrow = hN + b * H_;
  float acc = 0.f;
  #pragma unroll 8
  for (int h = 0; h < H_; ++h) acc = fmaf(hrow[h], W2[h * U_ + u], acc);
  hb[idx] = acc + W2b[u] + W1b[u];
}

// ---------------- Kernel 3: fused GEMM + tanh + V-dot ---------------------
// Grid: (B*S)/32 workgroups of 256 threads (8 waves).
// Each WG: 32 encoder rows x all 1024 U columns; wave w owns cols [128w,128w+128).
#define KC   512                 // K chunk staged in LDS
#define KPAD (KC + 8)            // padded stride (bf16 elems) to kill bank conflicts

__global__ __launch_bounds__(256)
void attn_score_kernel(const float* __restrict__ enc,
                       const unsigned short* __restrict__ W1T,
                       const float* __restrict__ hb,
                       const float* __restrict__ Vw,
                       float* __restrict__ score) {
  __shared__ __align__(16) unsigned short ldsA[32 * KPAD];   // ~33 KB
  __shared__ float ldsPart[8][32];

  const int tid  = threadIdx.x;
  const int lane = tid & 31;
  const int wave = tid >> 5;
  const int g    = lane >> 4;        // half-wave group
  const int ln   = lane & 15;
  const int nBase = wave * 128;

  const int rowBase = blockIdx.x * 32;     // global (b*s) row
  const int b = rowBase >> 11;             // / S_
  const float* src = enc + (size_t)rowBase * H_;

  v8f acc[2][8];
  #pragma unroll
  for (int mt = 0; mt < 2; ++mt)
    #pragma unroll
    for (int nt = 0; nt < 8; ++nt)
      acc[mt][nt] = (v8f){0.f, 0.f, 0.f, 0.f, 0.f, 0.f, 0.f, 0.f};

  union Frag { v16bf v; uint4 q[2]; };

  for (int kc = 0; kc < H_; kc += KC) {
    __syncthreads();   // all waves done reading previous chunk
    // stage 32 rows x KC cols, f32 -> bf16 (128 float4 per row)
    for (int i = tid; i < 32 * (KC / 4); i += 256) {
      int row = i >> 7;
      int c4  = i & 127;
      float4 f = reinterpret_cast<const float4*>(src + row * H_ + kc)[c4];
      ushort4 p;
      p.x = f32_to_bf16_rne(f.x);
      p.y = f32_to_bf16_rne(f.y);
      p.z = f32_to_bf16_rne(f.z);
      p.w = f32_to_bf16_rne(f.w);
      *reinterpret_cast<ushort4*>(&ldsA[row * KPAD + (c4 << 2)]) = p;
    }
    __syncthreads();

    #pragma unroll 1
    for (int k0 = 0; k0 < KC; k0 += 32) {
      const int kOff = k0 + 8 * g;              // local LDS k offset
      const int kAbs = kc + kOff;               // absolute k for W1T
      Frag a0, a1;
      a0.q[0] = *reinterpret_cast<const uint4*>(&ldsA[ln        * KPAD + kOff]);
      a0.q[1] = *reinterpret_cast<const uint4*>(&ldsA[ln        * KPAD + kOff + 16]);
      a1.q[0] = *reinterpret_cast<const uint4*>(&ldsA[(16 + ln) * KPAD + kOff]);
      a1.q[1] = *reinterpret_cast<const uint4*>(&ldsA[(16 + ln) * KPAD + kOff + 16]);
      #pragma unroll
      for (int nt = 0; nt < 8; ++nt) {
        const int n = nBase + nt * 16 + ln;
        const unsigned short* wp = W1T + (size_t)n * H_ + kAbs;
        Frag bf;
        bf.q[0] = *reinterpret_cast<const uint4*>(wp);
        bf.q[1] = *reinterpret_cast<const uint4*>(wp + 16);
        acc[0][nt] = __builtin_amdgcn_wmma_f32_16x16x32_bf16(
            false, a0.v, false, bf.v, (short)0, acc[0][nt], false, false);
        acc[1][nt] = __builtin_amdgcn_wmma_f32_16x16x32_bf16(
            false, a1.v, false, bf.v, (short)0, acc[1][nt], false, false);
      }
    }
  }

  // Epilogue: score contribution = sum_n tanh(acc + hb[b,n]) * Vw[n]
  float hbv[8], vwv[8];
  #pragma unroll
  for (int nt = 0; nt < 8; ++nt) {
    int n = nBase + nt * 16 + ln;
    hbv[nt] = hb[b * U_ + n];
    vwv[nt] = Vw[n];
  }
  #pragma unroll
  for (int mt = 0; mt < 2; ++mt) {
    #pragma unroll
    for (int r = 0; r < 8; ++r) {
      float part = 0.f;
      #pragma unroll
      for (int nt = 0; nt < 8; ++nt)
        part += tanhf(acc[mt][nt][r] + hbv[nt]) * vwv[nt];
      // butterfly over the 16-lane half (D layout: lane = N column)
      #pragma unroll
      for (int m = 1; m < 16; m <<= 1)
        part += __shfl_xor(part, m, 32);
      if (ln == 0)
        ldsPart[wave][mt * 16 + r + 8 * g] = part;   // unique (wave,row) writer
    }
  }
  __syncthreads();
  if (tid < 32) {
    float s = 0.f;
    #pragma unroll
    for (int w = 0; w < 8; ++w) s += ldsPart[w][tid];
    score[rowBase + tid] = s;    // V_b dropped: constant shift, softmax-invariant
  }
}

// ---------------- Kernel 4: softmax over S per batch ----------------------
__global__ __launch_bounds__(256)
void softmax_kernel(const float* __restrict__ score, float* __restrict__ out) {
  __shared__ float red[256];
  const int b = blockIdx.x;
  const int tid = threadIdx.x;
  const float* s = score + b * S_;
  float v[8];
  float mx = -INFINITY;
  #pragma unroll
  for (int i = 0; i < 8; ++i) { v[i] = s[tid + i * 256]; mx = fmaxf(mx, v[i]); }
  red[tid] = mx; __syncthreads();
  for (int off = 128; off > 0; off >>= 1) {
    if (tid < off) red[tid] = fmaxf(red[tid], red[tid + off]);
    __syncthreads();
  }
  mx = red[0]; __syncthreads();
  float sum = 0.f;
  #pragma unroll
  for (int i = 0; i < 8; ++i) { v[i] = __expf(v[i] - mx); sum += v[i]; }
  red[tid] = sum; __syncthreads();
  for (int off = 128; off > 0; off >>= 1) {
    if (tid < off) red[tid] += red[tid + off];
    __syncthreads();
  }
  const float inv = 1.f / red[0];
  #pragma unroll
  for (int i = 0; i < 8; ++i) out[b * S_ + tid + i * 256] = v[i] * inv;
}

// --------------------------------------------------------------------------
extern "C" void kernel_launch(void* const* d_in, const int* in_sizes, int n_in,
                              void* d_out, int out_size, void* d_ws, size_t ws_size,
                              hipStream_t stream) {
  const float* enc  = (const float*)d_in[0];   // [B,S,H]
  const float* hN   = (const float*)d_in[1];   // [B,H]
  const float* W1   = (const float*)d_in[2];   // [H,U]
  const float* W1b  = (const float*)d_in[3];   // [U]
  const float* W2   = (const float*)d_in[4];   // [H,U]
  const float* W2b  = (const float*)d_in[5];   // [U]
  const float* Vw   = (const float*)d_in[6];   // [U,1]
  // d_in[7] = V_b : constant shift, softmax-invariant -> unused

  // workspace layout
  unsigned short* W1T = (unsigned short*)d_ws;                       // 2 MB
  float* hb  = (float*)((char*)d_ws + (size_t)2 * 1024 * 1024);      // 128 KB
  float* sc  = (float*)((char*)hb   + (size_t)B_ * U_ * sizeof(float)); // 256 KB

  w1_transpose_bf16<<<(H_ * U_) / 256, 256, 0, stream>>>(W1, W1T);
  combined_bias_kernel<<<(B_ * U_) / 256, 256, 0, stream>>>(hN, W2, W2b, W1b, hb);
  attn_score_kernel<<<(B_ * S_) / 32, 256, 0, stream>>>(enc, W1T, hb, Vw, sc);
  softmax_kernel<<<B_, 256, 0, stream>>>(sc, (float*)d_out);
}
